// LinearAttention_13254269076194
// MI455X (gfx1250) — matrix operand order repaired and checked by reading, not compiled, not run on hardware
//
#include <hip/hip_runtime.h>
#include <hip/hip_bf16.h>

// ---------------------------------------------------------------------------
// CDNA5 (gfx1250) chunked linear attention, bf16 WMMA pipeline.
// wave32, 256-thread blocks (8 waves), v_wmma_f32_16x16x32_bf16 for all
// matrix ops, GLOBAL_LOAD_ASYNC_TO_LDS_B128 (ASYNCcnt) for GEMM staging.
// ---------------------------------------------------------------------------

typedef unsigned short u16;
typedef __attribute__((ext_vector_type(16))) __bf16 v16bf;
typedef __attribute__((ext_vector_type(8)))  float  v8f;

union Frag { v16bf v; unsigned int u[8]; };
union Pack { uint4 q; u16 s[8]; };

#define B_SZ 2
#define T_SZ 8192
#define HID_ 1024
#define H_   8
#define DK_  128
#define NCH  128   // chunks per sequence
#define CH   64    // chunk length

__device__ __forceinline__ u16 f2bf(float f) {
  unsigned int u = __float_as_uint(f);
  u += 0x7FFFu + ((u >> 16) & 1u);          // round-to-nearest-even
  return (u16)(u >> 16);
}
__device__ __forceinline__ float bf2f(u16 h) {
  return __uint_as_float(((unsigned int)h) << 16);
}

// --- CDNA5 async global->LDS copy (per-lane 16B), tracked by ASYNCcnt ------
// LDS aperture has zero low 32 bits, so truncating a generic pointer to a
// __shared__ object yields the DS byte address the instruction expects.
__device__ __forceinline__ void async_b128(const void* gaddr, void* lds) {
  unsigned loff = (unsigned)(size_t)lds;
  asm volatile("global_load_async_to_lds_b128 %0, %1, off"
               :: "v"(loff), "v"(gaddr) : "memory");
}
__device__ __forceinline__ void wait_async0() {
  asm volatile("s_wait_asynccnt 0" ::: "memory");
}

// ---------------------------------------------------------------------------
// fp32 -> bf16 conversion
// ---------------------------------------------------------------------------
__global__ void k_f32_to_bf16(const float* __restrict__ src,
                              u16* __restrict__ dst, int n) {
  int i = blockIdx.x * blockDim.x + threadIdx.x;
  if (i < n) dst[i] = f2bf(src[i]);
}

// ---------------------------------------------------------------------------
// Generic bf16 GEMM: C[M,N] = A[M,K] @ W[N,K]^T  (torch Linear convention)
// 128x128 tile per workgroup, K-step 32 (one WMMA per stage), 8 waves,
// wave = 64x32 patch. Double-buffered LDS, async-to-LDS staging overlapped
// with WMMA compute; one barrier per K-step.
// ---------------------------------------------------------------------------
__global__ __launch_bounds__(256) void k_gemm128(
    const u16* __restrict__ A, const u16* __restrict__ W,
    u16* __restrict__ Db, float* __restrict__ Df,
    int M, int N, int K, int store_f32)
{
  __shared__ u16 As[2][128 * 40];
  __shared__ u16 Bs[2][128 * 40];

  const int tid  = threadIdx.x;
  const int lane = tid & 31, wid = tid >> 5;
  const int g = lane >> 4, g4 = g * 4, l15 = lane & 15;
  const int wm = wid >> 2, wn = wid & 3;           // 2 x 4 wave grid
  const int mB = blockIdx.y * 128, nB = blockIdx.x * 128;
  const int r = tid >> 1, half = tid & 1;

  const u16* gA = A + (size_t)(mB + r) * K + half * 16;   // this thread's A src
  const u16* gW = W + (size_t)(nB + r) * K + half * 16;   // this thread's W src
  const int ldsOff = r * 40 + half * 16;

  v8f c[4][2] = {};

  // prologue: stage tile 0 into buffer 0
  async_b128(gA,     &As[0][ldsOff]);
  async_b128(gA + 8, &As[0][ldsOff + 8]);
  async_b128(gW,     &Bs[0][ldsOff]);
  async_b128(gW + 8, &Bs[0][ldsOff + 8]);

  const int niter = K >> 5;
  for (int i = 0; i < niter; ++i) {
    const int cur = i & 1;
    wait_async0();        // this wave's copies into buf[cur] are done
    __syncthreads();      // all waves' copies visible; buf[cur^1] free

    if (i + 1 < niter) {  // stage next tile while computing this one
      const u16* nA = gA + (size_t)(i + 1) * 32;
      const u16* nW = gW + (size_t)(i + 1) * 32;
      const int nb = cur ^ 1;
      async_b128(nA,     &As[nb][ldsOff]);
      async_b128(nA + 8, &As[nb][ldsOff + 8]);
      async_b128(nW,     &Bs[nb][ldsOff]);
      async_b128(nW + 8, &Bs[nb][ldsOff + 8]);
    }

    Frag a[4], bf[2];
#pragma unroll
    for (int t = 0; t < 4; ++t) {
      const unsigned int* ap = (const unsigned int*)&As[cur][(wm * 64 + t * 16 + l15) * 40];
#pragma unroll
      for (int v = 0; v < 4; ++v) { a[t].u[v] = ap[g4 + v]; a[t].u[v + 4] = ap[8 + g4 + v]; }
    }
#pragma unroll
    for (int j = 0; j < 2; ++j) {
      const unsigned int* bp = (const unsigned int*)&Bs[cur][(wn * 32 + j * 16 + l15) * 40];
#pragma unroll
      for (int v = 0; v < 8; ++v) bf[j].u[v] = bp[((v >> 2) << 3) + g4 + (v & 3)];
    }
#pragma unroll
    for (int t = 0; t < 4; ++t)
#pragma unroll
      for (int j = 0; j < 2; ++j)
        c[t][j] = __builtin_amdgcn_wmma_f32_16x16x32_bf16(
            false, a[t].v, false, bf[j].v, (short)0, c[t][j], false, false);
  }

#pragma unroll
  for (int t = 0; t < 4; ++t)
    for (int j = 0; j < 2; ++j)
#pragma unroll
      for (int v = 0; v < 8; ++v) {
        int row = mB + wm * 64 + t * 16 + v + 8 * g;
        int col = nB + wn * 32 + j * 16 + l15;
        float val = c[t][j][v];
        if (store_f32) Df[(size_t)row * N + col] = val;
        else           Db[(size_t)row * N + col] = f2bf(val);
      }
}

// ---------------------------------------------------------------------------
// Hadamard feature map (in place): Q[128 rows, head h] =
//   (Q @ W1^T + b1) * (Q @ W2^T + b2) * scale
// ---------------------------------------------------------------------------
__global__ __launch_bounds__(256) void k_feature_map(
    u16* __restrict__ Q, const u16* __restrict__ W1, const u16* __restrict__ W2,
    const float* __restrict__ bias1, const float* __restrict__ bias2, float scale)
{
  extern __shared__ u16 sm_fm[];
  u16* At  = sm_fm;             // 128*136
  u16* W1t = At  + 128 * 136;   // 128*136
  u16* W2t = W1t + 128 * 136;   // 128*136

  const int tid  = threadIdx.x;
  const int lane = tid & 31, wid = tid >> 5;
  const int g = lane >> 4, g4 = g * 4, l15 = lane & 15;
  const int rB = blockIdx.x * 128;
  const int h  = blockIdx.y;
  const int r = tid >> 1, half = tid & 1;

  { // stage activation block + both weight matrices via async DMA
    const u16* gq = Q  + (size_t)(rB + r) * HID_ + h * DK_ + half * 64;
    const u16* g1 = W1 + (size_t)r * DK_ + half * 64;
    const u16* g2 = W2 + (size_t)r * DK_ + half * 64;
#pragma unroll
    for (int i = 0; i < 8; ++i) {
      async_b128(gq + i * 8, &At [r * 136 + half * 64 + i * 8]);
      async_b128(g1 + i * 8, &W1t[r * 136 + half * 64 + i * 8]);
      async_b128(g2 + i * 8, &W2t[r * 136 + half * 64 + i * 8]);
    }
  }
  wait_async0();
  __syncthreads();

  Frag a[4];  // A fragments for all 4 K-steps (m-tile = wave id)
#pragma unroll
  for (int ks = 0; ks < 4; ++ks) {
    const unsigned int* ap = (const unsigned int*)&At[(wid * 16 + l15) * 136];
#pragma unroll
    for (int v = 0; v < 4; ++v) {
      a[ks].u[v]     = ap[ks * 16 + g4 + v];
      a[ks].u[v + 4] = ap[ks * 16 + 8 + g4 + v];
    }
  }

  for (int nt = 0; nt < 8; ++nt) {
    v8f c1 = {}, c2 = {};
#pragma unroll
    for (int ks = 0; ks < 4; ++ks) {
      Frag b1f, b2f;
      const unsigned int* b1p = (const unsigned int*)&W1t[(nt * 16 + l15) * 136];
      const unsigned int* b2p = (const unsigned int*)&W2t[(nt * 16 + l15) * 136];
#pragma unroll
      for (int v = 0; v < 8; ++v) {
        b1f.u[v] = b1p[ks * 16 + ((v >> 2) << 3) + g4 + (v & 3)];
        b2f.u[v] = b2p[ks * 16 + ((v >> 2) << 3) + g4 + (v & 3)];
      }
      c1 = __builtin_amdgcn_wmma_f32_16x16x32_bf16(false, a[ks].v, false, b1f.v, (short)0, c1, false, false);
      c2 = __builtin_amdgcn_wmma_f32_16x16x32_bf16(false, a[ks].v, false, b2f.v, (short)0, c2, false, false);
    }
    int colh = nt * 16 + l15;
    float bb1 = bias1[colh], bb2 = bias2[colh];
#pragma unroll
    for (int v = 0; v < 8; ++v) {
      int row = rB + wid * 16 + v + 8 * g;
      float val = (c1[v] + bb1) * (c2[v] + bb2) * scale;
      Q[(size_t)row * HID_ + h * DK_ + colh] = f2bf(val);
    }
  }
}

// ---------------------------------------------------------------------------
// Per-chunk kv sums: KV[b,n,h] = k_c^T @ v_c   (128x128 from 64 tokens)
// ---------------------------------------------------------------------------
__global__ __launch_bounds__(256) void k_chunk_kv(
    const u16* __restrict__ Kb, const u16* __restrict__ Vb, u16* __restrict__ KV)
{
  __shared__ u16 kT[128 * 72];  // [dk][token]
  __shared__ u16 vT[128 * 72];  // [dv][token]

  const int tid  = threadIdx.x;
  const int lane = tid & 31, wid = tid >> 5;
  const int g = lane >> 4, g4 = g * 4, l15 = lane & 15;
  const int h = blockIdx.x, n = blockIdx.y, b = blockIdx.z;
  const int t0 = n * CH;

  { // stage transposed tiles (element transpose -> manual path)
    const int t = tid & 63, seg = tid >> 6;
    const u16* gk = Kb + (size_t)(b * T_SZ + t0 + t) * HID_ + h * DK_ + seg * 32;
    const u16* gv = Vb + (size_t)(b * T_SZ + t0 + t) * HID_ + h * DK_ + seg * 32;
    Pack pk[4], pv[4];
#pragma unroll
    for (int i = 0; i < 4; ++i) { pk[i].q = ((const uint4*)gk)[i]; pv[i].q = ((const uint4*)gv)[i]; }
#pragma unroll
    for (int i = 0; i < 4; ++i)
#pragma unroll
      for (int e = 0; e < 8; ++e) {
        int d = seg * 32 + i * 8 + e;
        kT[d * 72 + t] = pk[i].s[e];
        vT[d * 72 + t] = pv[i].s[e];
      }
  }
  __syncthreads();

  const int wm = wid >> 2, wn = wid & 3;
  v8f c[4][2] = {};
#pragma unroll
  for (int ks = 0; ks < 2; ++ks) {
    Frag a[4], bf[2];
#pragma unroll
    for (int t = 0; t < 4; ++t) {
      const unsigned int* ap = (const unsigned int*)&kT[(wm * 64 + t * 16 + l15) * 72];
#pragma unroll
      for (int v = 0; v < 4; ++v) {
        a[t].u[v]     = ap[ks * 16 + g4 + v];
        a[t].u[v + 4] = ap[ks * 16 + 8 + g4 + v];
      }
    }
#pragma unroll
    for (int j = 0; j < 2; ++j) {
      const unsigned int* bp = (const unsigned int*)&vT[(wn * 32 + j * 16 + l15) * 72];
#pragma unroll
      for (int v = 0; v < 8; ++v) bf[j].u[v] = bp[ks * 16 + ((v >> 2) << 3) + g4 + (v & 3)];
    }
#pragma unroll
    for (int t = 0; t < 4; ++t)
#pragma unroll
      for (int j = 0; j < 2; ++j)
        c[t][j] = __builtin_amdgcn_wmma_f32_16x16x32_bf16(
            false, a[t].v, false, bf[j].v, (short)0, c[t][j], false, false);
  }

  u16* out = KV + ((size_t)((b * NCH + n) * H_ + h) << 14);
#pragma unroll
  for (int t = 0; t < 4; ++t)
    for (int j = 0; j < 2; ++j)
#pragma unroll
      for (int v = 0; v < 8; ++v) {
        int row = wm * 64 + t * 16 + v + 8 * g;      // dk
        int col = wn * 32 + j * 16 + l15;            // dv
        out[row * 128 + col] = f2bf(c[t][j][v]);
      }
}

// ---------------------------------------------------------------------------
// In-place exclusive prefix-sum over chunks (state entering each chunk).
// L2-resident: the whole kv array is 64 MB bf16 << 192 MB L2.
// ---------------------------------------------------------------------------
__global__ void k_cumsum_excl(u16* __restrict__ KV) {
  int t = blockIdx.x * blockDim.x + threadIdx.x;   // 0 .. B*H*16384-1
  int e = t & 16383;
  int h = (t >> 14) & (H_ - 1);
  int b = t >> 17;
  float acc = 0.f;
  for (int n = 0; n < NCH; ++n) {
    size_t idx = ((size_t)((b * NCH + n) * H_ + h) << 14) + e;
    u16 cur = KV[idx];
    KV[idx] = f2bf(acc);
    acc += bf2f(cur);
  }
}

// ---------------------------------------------------------------------------
// Fused attention chunk: o = q@S + tril(q@k^T)@v, then RMSNorm * rms_w.
// One workgroup per (b, chunk, head).
// ---------------------------------------------------------------------------
__global__ __launch_bounds__(256) void k_attn(
    const u16* __restrict__ Qb, const u16* __restrict__ Kb, const u16* __restrict__ Vb,
    const u16* __restrict__ S, const float* __restrict__ rms_w, u16* __restrict__ Ob)
{
  extern __shared__ u16 sm[];
  u16* qt = sm;                    // [token][dk]  64*136
  u16* kt = qt + 64 * 136;         // [token][dk]  64*136
  u16* vT = kt + 64 * 136;         // [dv][token]  128*72
  u16* ST = vT + 128 * 72;         // [dv][dk]     128*136
  u16* At = ST + 128 * 136;        // [i][j]       64*72
  float* rowssq = (float*)(At + 64 * 72);  // 64 floats

  const int tid  = threadIdx.x;
  const int lane = tid & 31, wid = tid >> 5;
  const int g = lane >> 4, g4 = g * 4, l15 = lane & 15;
  const int h = blockIdx.x, n = blockIdx.y, b = blockIdx.z;
  const int t0 = n * CH;

  { // q,k natural layout -> async DMA; v transposed -> manual
    const int t = tid & 63, seg = tid >> 6;
    const u16* gq = Qb + (size_t)(b * T_SZ + t0 + t) * HID_ + h * DK_ + seg * 32;
    const u16* gk = Kb + (size_t)(b * T_SZ + t0 + t) * HID_ + h * DK_ + seg * 32;
    const u16* gv = Vb + (size_t)(b * T_SZ + t0 + t) * HID_ + h * DK_ + seg * 32;
    async_b128(gq,     &qt[t * 136 + seg * 32]);
    async_b128(gq + 8, &qt[t * 136 + seg * 32 + 8]);
    async_b128(gq + 16, &qt[t * 136 + seg * 32 + 16]);
    async_b128(gq + 24, &qt[t * 136 + seg * 32 + 24]);
    async_b128(gk,     &kt[t * 136 + seg * 32]);
    async_b128(gk + 8, &kt[t * 136 + seg * 32 + 8]);
    async_b128(gk + 16, &kt[t * 136 + seg * 32 + 16]);
    async_b128(gk + 24, &kt[t * 136 + seg * 32 + 24]);
    Pack pv[4];
#pragma unroll
    for (int i = 0; i < 4; ++i) pv[i].q = ((const uint4*)gv)[i];
#pragma unroll
    for (int i = 0; i < 4; ++i)
#pragma unroll
      for (int e = 0; e < 8; ++e)
        vT[(seg * 32 + i * 8 + e) * 72 + t] = pv[i].s[e];
  }
  { // S transposed -> ST[dv][dk]
    const int dk = tid >> 1, half = tid & 1;
    const u16* gs = S + ((size_t)((b * NCH + n) * H_ + h) << 14) + dk * 128 + half * 64;
    Pack ps[8];
#pragma unroll
    for (int i = 0; i < 8; ++i) ps[i].q = ((const uint4*)gs)[i];
#pragma unroll
    for (int i = 0; i < 8; ++i)
#pragma unroll
      for (int e = 0; e < 8; ++e)
        ST[(half * 64 + i * 8 + e) * 136 + dk] = ps[i].s[e];
  }
  if (tid < 64) rowssq[tid] = 0.f;
  wait_async0();
  __syncthreads();

  // --- GEMM2: A = q @ k^T (64x64), causal mask, store bf16 to LDS ---
  {
    const int wm2 = wid & 3, wn2 = wid >> 2;
    v8f cA[2] = {};
#pragma unroll
    for (int ks = 0; ks < 4; ++ks) {
      Frag a;
      const unsigned int* ap = (const unsigned int*)&qt[(wm2 * 16 + l15) * 136];
#pragma unroll
      for (int v = 0; v < 4; ++v) {
        a.u[v]     = ap[ks * 16 + g4 + v];
        a.u[v + 4] = ap[ks * 16 + 8 + g4 + v];
      }
#pragma unroll
      for (int jj = 0; jj < 2; ++jj) {
        Frag bf;
        const unsigned int* bp = (const unsigned int*)&kt[((wn2 * 2 + jj) * 16 + l15) * 136];
#pragma unroll
        for (int v = 0; v < 8; ++v) bf.u[v] = bp[ks * 16 + ((v >> 2) << 3) + g4 + (v & 3)];
        cA[jj] = __builtin_amdgcn_wmma_f32_16x16x32_bf16(false, a.v, false, bf.v, (short)0, cA[jj], false, false);
      }
    }
#pragma unroll
    for (int jj = 0; jj < 2; ++jj)
#pragma unroll
      for (int v = 0; v < 8; ++v) {
        int i    = wm2 * 16 + v + 8 * g;
        int jcol = (wn2 * 2 + jj) * 16 + l15;
        float val = (jcol <= i) ? cA[jj][v] : 0.f;
        At[i * 72 + jcol] = f2bf(val);
      }
  }
  __syncthreads();

  // --- GEMM1 (q@S) + GEMM3 (A@v) accumulated: out 64x128 ---
  const int wm = wid & 3, wn = wid >> 2;
  v8f co[4] = {};
#pragma unroll
  for (int ks = 0; ks < 4; ++ks) {        // q @ S, K = 128
    Frag a;
    const unsigned int* ap = (const unsigned int*)&qt[(wm * 16 + l15) * 136];
#pragma unroll
    for (int v = 0; v < 4; ++v) {
      a.u[v]     = ap[ks * 16 + g4 + v];
      a.u[v + 4] = ap[ks * 16 + 8 + g4 + v];
    }
#pragma unroll
    for (int j = 0; j < 4; ++j) {
      Frag bf;
      const unsigned int* bp = (const unsigned int*)&ST[(wn * 64 + j * 16 + l15) * 136];
#pragma unroll
      for (int v = 0; v < 8; ++v) bf.u[v] = bp[ks * 16 + ((v >> 2) << 3) + g4 + (v & 3)];
      co[j] = __builtin_amdgcn_wmma_f32_16x16x32_bf16(false, a.v, false, bf.v, (short)0, co[j], false, false);
    }
  }
#pragma unroll
  for (int ks = 0; ks < 2; ++ks) {        // A @ v, K = 64
    Frag a;
    const unsigned int* ap = (const unsigned int*)&At[(wm * 16 + l15) * 72];
#pragma unroll
    for (int v = 0; v < 4; ++v) {
      a.u[v]     = ap[ks * 16 + g4 + v];
      a.u[v + 4] = ap[ks * 16 + 8 + g4 + v];
    }
#pragma unroll
    for (int j = 0; j < 4; ++j) {
      Frag bf;
      const unsigned int* bp = (const unsigned int*)&vT[(wn * 64 + j * 16 + l15) * 72];
#pragma unroll
      for (int v = 0; v < 8; ++v) bf.u[v] = bp[ks * 16 + ((v >> 2) << 3) + g4 + (v & 3)];
      co[j] = __builtin_amdgcn_wmma_f32_16x16x32_bf16(false, a.v, false, bf.v, (short)0, co[j], false, false);
    }
  }

  // --- RMSNorm: row sums of squares via LDS atomics, then scale + store ---
#pragma unroll
  for (int v = 0; v < 8; ++v) {
    int rloc = wm * 16 + v + 8 * g;
    float ss = 0.f;
#pragma unroll
    for (int j = 0; j < 4; ++j) ss += co[j][v] * co[j][v];
    atomicAdd(&rowssq[rloc], ss);
  }
  __syncthreads();

#pragma unroll
  for (int v = 0; v < 8; ++v) {
    int rloc = wm * 16 + v + 8 * g;
    float inv = rsqrtf(rowssq[rloc] * (1.f / 128.f) + 1e-5f);
#pragma unroll
    for (int j = 0; j < 4; ++j) {
      int dv = wn * 64 + j * 16 + l15;
      float val = co[j][v] * inv * rms_w[dv];
      Ob[(size_t)(b * T_SZ + t0 + rloc) * HID_ + h * DK_ + dv] = f2bf(val);
    }
  }
}

// ---------------------------------------------------------------------------
// Host orchestration
// ---------------------------------------------------------------------------
extern "C" void kernel_launch(void* const* d_in, const int* in_sizes, int n_in,
                              void* d_out, int out_size, void* d_ws, size_t ws_size,
                              hipStream_t stream)
{
  const float* x      = (const float*)d_in[0];
  const float* Wq     = (const float*)d_in[1];
  const float* Wk     = (const float*)d_in[2];
  const float* Wv     = (const float*)d_in[3];
  const float* fmq_w1 = (const float*)d_in[4];
  const float* fmq_b1 = (const float*)d_in[5];
  const float* fmq_w2 = (const float*)d_in[6];
  const float* fmq_b2 = (const float*)d_in[7];
  const float* fmk_w1 = (const float*)d_in[8];
  const float* fmk_b1 = (const float*)d_in[9];
  const float* fmk_w2 = (const float*)d_in[10];
  const float* fmk_b2 = (const float*)d_in[11];
  const float* rms_w  = (const float*)d_in[12];
  const float* Wo     = (const float*)d_in[13];

  const size_t MTOK = (size_t)B_SZ * T_SZ;   // 16384 tokens

  u16* xb  = (u16*)d_ws;
  u16* qb  = xb  + MTOK * HID_;
  u16* kb  = qb  + MTOK * HID_;
  u16* vb  = kb  + MTOK * HID_;
  u16* kvS = vb  + MTOK * HID_;                          // B*NCH*H*128*128
  u16* wqb = kvS + (size_t)B_SZ * NCH * H_ * DK_ * DK_;
  u16* wkb = wqb + (size_t)HID_ * HID_;
  u16* wvb = wkb + (size_t)HID_ * HID_;
  u16* wob = wvb + (size_t)HID_ * HID_;
  u16* fq1 = wob + (size_t)HID_ * HID_;
  u16* fq2 = fq1 + DK_ * DK_;
  u16* fk1 = fq2 + DK_ * DK_;
  u16* fk2 = fk1 + DK_ * DK_;
  u16* ob  = xb;   // reuse x buffer after projections are done

  auto conv = [&](const float* s, u16* d, int nElem) {
    k_f32_to_bf16<<<dim3((nElem + 255) / 256), dim3(256), 0, stream>>>(s, d, nElem);
  };
  conv(x,  xb,  (int)(MTOK * HID_));
  conv(Wq, wqb, HID_ * HID_);
  conv(Wk, wkb, HID_ * HID_);
  conv(Wv, wvb, HID_ * HID_);
  conv(Wo, wob, HID_ * HID_);
  conv(fmq_w1, fq1, DK_ * DK_);
  conv(fmq_w2, fq2, DK_ * DK_);
  conv(fmk_w1, fk1, DK_ * DK_);
  conv(fmk_w2, fk2, DK_ * DK_);

  dim3 gg(HID_ / 128, (unsigned)(MTOK / 128));   // (8, 128)
  k_gemm128<<<gg, 256, 0, stream>>>(xb, wqb, qb, nullptr, (int)MTOK, HID_, HID_, 0);
  k_gemm128<<<gg, 256, 0, stream>>>(xb, wkb, kb, nullptr, (int)MTOK, HID_, HID_, 0);
  k_gemm128<<<gg, 256, 0, stream>>>(xb, wvb, vb, nullptr, (int)MTOK, HID_, HID_, 0);

  size_t fm_smem = (size_t)3 * 128 * 136 * sizeof(u16);
  k_feature_map<<<dim3((unsigned)(MTOK / 128), H_), 256, fm_smem, stream>>>(
      qb, fq1, fq2, fmq_b1, fmq_b2, 0.08838834764831845f /* 1/sqrt(128) */);
  k_feature_map<<<dim3((unsigned)(MTOK / 128), H_), 256, fm_smem, stream>>>(
      kb, fk1, fk2, fmk_b1, fmk_b2, 1.0f);

  k_chunk_kv<<<dim3(H_, NCH, B_SZ), 256, 0, stream>>>(kb, vb, kvS);
  k_cumsum_excl<<<dim3((B_SZ * H_ * DK_ * DK_) / 256), 256, 0, stream>>>(kvS);

  size_t attn_smem = (size_t)(64 * 136 + 64 * 136 + 128 * 72 + 128 * 136 + 64 * 72) * sizeof(u16)
                   + 64 * sizeof(float);
  k_attn<<<dim3(H_, NCH, B_SZ), 256, attn_smem, stream>>>(qb, kb, vb, kvS, rms_w, ob);

  k_gemm128<<<gg, 256, 0, stream>>>(ob, wob, nullptr, (float*)d_out, (int)MTOK, HID_, HID_, 1);
}
